// CrossMultiheadAttention_21715354649692
// MI455X (gfx1250) — matrix operand order repaired and checked by reading, not compiled, not run on hardware
//
#include <hip/hip_runtime.h>

typedef __attribute__((ext_vector_type(16))) _Float16 v16h;
typedef __attribute__((ext_vector_type(8)))  float    v8f;
typedef __attribute__((ext_vector_type(4)))  unsigned int u32x4;
typedef __attribute__((ext_vector_type(8)))  int      i32x8;
typedef __attribute__((ext_vector_type(4)))  int      i32x4;

union Frag16 {
  v16h     v;
  _Float16 h[16];
  uint4    q[2];
};

constexpr int kB = 4;
constexpr int kT = 1024;
constexpr int kS = 1024;
constexpr int kE = 1024;
constexpr int kH = 16;
constexpr int kD = 64;

// A-fragment (16x32 f16): lane row = lane&15; elems i at K = 16*half + i
// -> one contiguous 32B run per lane.
__device__ inline v16h load_a16(const _Float16* base, int half) {
  Frag16 f;
  f.q[0] = *(const uint4*)(base + 16 * half);
  f.q[1] = *(const uint4*)(base + 16 * half + 8);
  return f.v;
}

// B-fragment (32x16 f16): lane col = lane&15; elems i at K = 16*half + i
// -> one contiguous 32B run per lane.
__device__ inline v16h load_b16(const _Float16* base, int half) {
  Frag16 f;
  f.q[0] = *(const uint4*)(base + 16 * half);
  f.q[1] = *(const uint4*)(base + 16 * half + 8);
  return f.v;
}

// ---------------------------------------------------------------------------
// Tensor Data Mover: 2D tile load (global -> LDS), 16-bit elements.
// D# built per CDNA5 ISA (group0 128b, group1 256b).
// ---------------------------------------------------------------------------
__device__ inline void tdm_load_2d_f16(unsigned int lds_addr, const void* gaddr,
                                       unsigned int tile_d0,   // elems (contig)
                                       unsigned int tile_d1,   // rows
                                       unsigned int tensor_d0,
                                       unsigned int tensor_d1,
                                       unsigned int stride0) { // elems
  const unsigned long long ga = (unsigned long long)(uintptr_t)gaddr;
  u32x4 g0;
  g0[0] = 1u;                                       // count=1 (valid user D#)
  g0[1] = lds_addr;                                 // lds_addr [63:32]
  g0[2] = (unsigned int)(ga & 0xffffffffu);         // global_addr [95:64]
  g0[3] = (unsigned int)((ga >> 32) & 0x01ffffffu)  // global_addr [120:96]
          | 0x80000000u;                            // type=2 ("image") [127:126]
  i32x8 g1;
  g1[0] = (int)(1u << 16);                          // data_size=1 (2B); wg_mask=0
  g1[1] = (int)(tensor_d0 << 16);                   // tensor_dim0[15:0] @ 63:48
  g1[2] = (int)((tensor_d0 >> 16) | (tensor_d1 << 16));
  g1[3] = (int)((tensor_d1 >> 16) | (tile_d0 << 16));   // tile_dim0 @ 127:112
  g1[4] = (int)(tile_d1 & 0xffffu);                 // tile_dim1; tile_dim2=0
  g1[5] = (int)stride0;                             // tensor_dim0_stride[31:0]
  g1[6] = 0;                                        // stride hi / dim1_stride
  g1[7] = 0;
  const i32x4 z4 = {0, 0, 0, 0};
#if defined(__clang_major__) && __clang_major__ >= 23
  const i32x8 z8 = {0, 0, 0, 0, 0, 0, 0, 0};
  __builtin_amdgcn_tensor_load_to_lds(g0, g1, z4, z4, z8, 0);
#else
  __builtin_amdgcn_tensor_load_to_lds(g0, g1, z4, z4, 0);
#endif
}

// ---------------------------------------------------------------------------
// One-shot f32 -> f16 conversion (activations + weights). ~96MB total traffic,
// removes all v_cvt from the GEMM inner loops.
// ---------------------------------------------------------------------------
__global__ __launch_bounds__(256) void cvt_kernel(const float* __restrict__ src,
                                                  _Float16* __restrict__ dst,
                                                  int n) {
  const int i = (blockIdx.x * 256 + threadIdx.x) * 4;
  if (i < n) {
    const float4 x = *(const float4*)(src + i);
    _Float16 h4[4] = {(_Float16)x.x, (_Float16)x.y, (_Float16)x.z, (_Float16)x.w};
    *(uint2*)(dst + i) = *(const uint2*)h4;
  }
}

// ---------------------------------------------------------------------------
// Projection GEMM (f16 in, fp32 acc): one wave computes a 16x64 tile.
// Column group of 64 == exactly one head (h = blockIdx.y).
// ---------------------------------------------------------------------------
__global__ __launch_bounds__(32) void proj_kernel(
    const _Float16* __restrict__ X,   // [B*T, E] f16
    const _Float16* __restrict__ W,   // [E, E]   f16 (row = output feature)
    const float* __restrict__ bias,   // [E]      fp32
    _Float16* __restrict__ out,       // [B*H, T, D]
    float scale) {
  const int rt = blockIdx.x;   // 16-row tile
  const int h  = blockIdx.y;   // head == 64-col group
  const int lane = threadIdx.x;
  const int half = lane >> 4, ln = lane & 15;

  const _Float16* xrow = X + (size_t)(rt * 16 + ln) * kE;
  const _Float16* wrow[4];
#pragma unroll
  for (int j = 0; j < 4; ++j)
    wrow[j] = W + (size_t)(h * 64 + j * 16 + ln) * kE;

  v8f c[4] = {};
  for (int k0 = 0; k0 < kE; k0 += 32) {
    const v16h a = load_a16(xrow + k0, half);
#pragma unroll
    for (int j = 0; j < 4; ++j) {
      const v16h bf = load_b16(wrow[j] + k0, half);
      c[j] = __builtin_amdgcn_wmma_f32_16x16x32_f16(false, a, false, bf,
                                                    (short)0, c[j], false, false);
    }
  }

  // 16-row tile never crosses a batch boundary: b / t_base are uniform.
  const int b  = (rt * 16) >> 10;
  const int tb = (rt * 16) & 1023;
  _Float16* ob = out + (((size_t)b * kH + h) * kT + tb) * kD;
#pragma unroll
  for (int j = 0; j < 4; ++j) {
    const float bb = bias[h * 64 + j * 16 + ln];
#pragma unroll
    for (int r = 0; r < 8; ++r)
      ob[(size_t)(r + 8 * half) * kD + j * 16 + ln] =
          (_Float16)((c[j][r] + bb) * scale);
  }
}

// ---------------------------------------------------------------------------
// Flash attention: 4 waves/block, each wave owns a 16-row T tile of one (b,h).
// K/V 32x64 f16 chunks staged in LDS by the Tensor Data Mover (TENSORcnt),
// online softmax fp32, attn_bias stream prefetched (global_prefetch_b8).
// ---------------------------------------------------------------------------
__global__ __launch_bounds__(128) void attn_kernel(
    const _Float16* __restrict__ Q,       // [B*H, T, D]
    const _Float16* __restrict__ K,       // [B*H, S, D]
    const _Float16* __restrict__ V,       // [B*H, S, D]
    const unsigned char* __restrict__ mask,  // [B, S] bool
    const float* __restrict__ bias,       // [B*H, T, S]
    _Float16* __restrict__ O)             // [B, T, E] (= [B,T,H,D])
{
  __shared__ __align__(32) _Float16 Kt[32 * kD];
  __shared__ __align__(32) _Float16 Vt[32 * kD];
  __shared__ __align__(32) _Float16 Pt[4][16 * 32];

  const int bh = blockIdx.x;
  const int b = bh / kH, h = bh % kH;
  const int tid = threadIdx.x;
  const int w = tid >> 5;
  const int lane = tid & 31;
  const int half = lane >> 4, ln = lane & 15;
  const int t0 = blockIdx.y * 64 + w * 16;

  const unsigned int ldsK = (unsigned int)(uintptr_t)&Kt[0];
  const unsigned int ldsV = (unsigned int)(uintptr_t)&Vt[0];

  // Q A-fragments (D=64 -> two 32-wide K chunks), Q already scaled by D^-1/2.
  const _Float16* qrow = Q + ((size_t)bh * kT + (t0 + ln)) * kD;
  const v16h aq0 = load_a16(qrow, half);
  const v16h aq1 = load_a16(qrow + 32, half);

  float m[8], l[8];
  v8f oc[4] = {};
#pragma unroll
  for (int i = 0; i < 8; ++i) { m[i] = -3.0e38f; l[i] = 0.0f; }

  const float* brow = bias + (size_t)bh * kT * kS;

  for (int s0 = 0; s0 < kS; s0 += 32) {
    // TDM: one wave issues both tile DMAs; s_wait_tensorcnt publishes to LDS,
    // workgroup barrier publishes to the other waves.
    if (tid == 0) {
      const size_t gofs = ((size_t)bh * kS + s0) * kD;
      tdm_load_2d_f16(ldsK, K + gofs, kD, 32, kD, 32, kD);
      tdm_load_2d_f16(ldsV, V + gofs, kD, 32, kD, 32, kD);
      __builtin_amdgcn_s_wait_tensorcnt(0);
    }
    __syncthreads();

    // scores: two 16x16 tiles (cols s0..s0+15, s0+16..s0+31), K over d.
    v8f c0 = {}, c1 = {};
    {
      const v16h b00 = *(const v16h*)&Kt[(0 * 16 + ln) * kD + 0  + 16 * half];
      const v16h b01 = *(const v16h*)&Kt[(0 * 16 + ln) * kD + 32 + 16 * half];
      const v16h b10 = *(const v16h*)&Kt[(1 * 16 + ln) * kD + 0  + 16 * half];
      const v16h b11 = *(const v16h*)&Kt[(1 * 16 + ln) * kD + 32 + 16 * half];
      c0 = __builtin_amdgcn_wmma_f32_16x16x32_f16(false, aq0, false, b00, (short)0, c0, false, false);
      c0 = __builtin_amdgcn_wmma_f32_16x16x32_f16(false, aq1, false, b01, (short)0, c0, false, false);
      c1 = __builtin_amdgcn_wmma_f32_16x16x32_f16(false, aq0, false, b10, (short)0, c1, false, false);
      c1 = __builtin_amdgcn_wmma_f32_16x16x32_f16(false, aq1, false, b11, (short)0, c1, false, false);
    }

    const int sa = s0 + ln, sb = s0 + 16 + ln;
    const bool ma = mask[b * kS + sa] != 0;
    const bool mb = mask[b * kS + sb] != 0;

    float p0[8], p1[8], alr[8];
#pragma unroll
    for (int r = 0; r < 8; ++r) {
      const int t = t0 + r + 8 * half;  // C layout row
      // prefetch next chunk's bias line for this row (the 256MB stream)
      __builtin_prefetch(&brow[(size_t)t * kS + sa + 32], 0, 0);
      float x0 = ma ? -3.0e38f : (c0[r] + brow[(size_t)t * kS + sa]);
      float x1 = mb ? -3.0e38f : (c1[r] + brow[(size_t)t * kS + sb]);
      float mx = fmaxf(x0, x1);
#pragma unroll
      for (int off = 1; off < 16; off <<= 1)
        mx = fmaxf(mx, __shfl_xor(mx, off, 16));
      const float mn = fmaxf(m[r], mx);
      const float al = __expf(m[r] - mn);
      const float e0 = ma ? 0.0f : __expf(x0 - mn);
      const float e1 = mb ? 0.0f : __expf(x1 - mn);
      float rs = e0 + e1;
#pragma unroll
      for (int off = 1; off < 16; off <<= 1) rs += __shfl_xor(rs, off, 16);
      l[r] = l[r] * al + rs;
      m[r] = mn;
      p0[r] = e0; p1[r] = e1; alr[r] = al;
    }

#pragma unroll
    for (int dt = 0; dt < 4; ++dt)
#pragma unroll
      for (int r = 0; r < 8; ++r) oc[dt][r] *= alr[r];

    // P: C-layout -> A-layout via this wave's LDS slice.
#pragma unroll
    for (int r = 0; r < 8; ++r) {
      const int rr = r + 8 * half;
      Pt[w][rr * 32 + ln]      = (_Float16)p0[r];
      Pt[w][rr * 32 + 16 + ln] = (_Float16)p1[r];
    }
    asm volatile("s_wait_dscnt 0" ::: "memory");  // LDS RAW within wave
    const v16h ap = load_a16(&Pt[w][ln * 32], half);

    // O += P @ V : B-frag of V, K = s (32), N = d (4 tiles of 16).
#pragma unroll
    for (int dt = 0; dt < 4; ++dt) {
      Frag16 bv;
      const int d = dt * 16 + ln;
#pragma unroll
      for (int i = 0; i < 16; ++i) bv.h[i] = Vt[(i + 16 * half) * kD + d];
      oc[dt] = __builtin_amdgcn_wmma_f32_16x16x32_f16(false, ap, false, bv.v,
                                                      (short)0, oc[dt], false, false);
    }
    __syncthreads();  // before next chunk's TDM overwrites Kt/Vt
  }

  // normalize + store f16 O in [B, T, H*D] (row layout for the output GEMM)
  _Float16* orow = O + (size_t)b * kT * kE + (size_t)h * kD;
#pragma unroll
  for (int r = 0; r < 8; ++r) {
    const int t = t0 + r + 8 * half;
    const float inv = 1.0f / l[r];
#pragma unroll
    for (int dt = 0; dt < 4; ++dt)
      orow[(size_t)t * kE + dt * 16 + ln] = (_Float16)(oc[dt][r] * inv);
  }
}

// ---------------------------------------------------------------------------
// Output projection: one wave computes a 16x64 fp32 tile of d_out.
// ---------------------------------------------------------------------------
__global__ __launch_bounds__(32) void outproj_kernel(
    const _Float16* __restrict__ A,   // [B*T, E] f16
    const _Float16* __restrict__ W,   // [E, E]   f16 (wo)
    const float* __restrict__ bias,   // [E]
    float* __restrict__ out) {        // [B*T, E] fp32
  const int rt = blockIdx.x, ng = blockIdx.y;
  const int lane = threadIdx.x;
  const int half = lane >> 4, ln = lane & 15;

  const _Float16* arow = A + (size_t)(rt * 16 + ln) * kE;
  const _Float16* wrow[4];
#pragma unroll
  for (int j = 0; j < 4; ++j)
    wrow[j] = W + (size_t)(ng * 64 + j * 16 + ln) * kE;

  v8f c[4] = {};
  for (int k0 = 0; k0 < kE; k0 += 32) {
    const v16h a = load_a16(arow + k0, half);
#pragma unroll
    for (int j = 0; j < 4; ++j) {
      const v16h bf = load_b16(wrow[j] + k0, half);
      c[j] = __builtin_amdgcn_wmma_f32_16x16x32_f16(false, a, false, bf,
                                                    (short)0, c[j], false, false);
    }
  }

  float* ob = out + (size_t)(rt * 16) * kE + ng * 64;
#pragma unroll
  for (int j = 0; j < 4; ++j) {
    const float bb = bias[ng * 64 + j * 16 + ln];
#pragma unroll
    for (int r = 0; r < 8; ++r)
      ob[(size_t)(r + 8 * half) * kE + j * 16 + ln] = c[j][r] + bb;
  }
}

extern "C" void kernel_launch(void* const* d_in, const int* in_sizes, int n_in,
                              void* d_out, int out_size, void* d_ws, size_t ws_size,
                              hipStream_t stream) {
  (void)in_sizes; (void)n_in; (void)out_size; (void)ws_size;
  const float* query = (const float*)d_in[0];
  const float* key_  = (const float*)d_in[1];
  const float* value = (const float*)d_in[2];
  const unsigned char* mask = (const unsigned char*)d_in[3];  // bool [B,S]
  const float* bias = (const float*)d_in[4];                  // [B*H, T, S]
  const float* wq = (const float*)d_in[5];
  const float* bq = (const float*)d_in[6];
  const float* wk = (const float*)d_in[7];
  const float* bk = (const float*)d_in[8];
  const float* wv = (const float*)d_in[9];
  const float* bv = (const float*)d_in[10];
  const float* wo = (const float*)d_in[11];
  const float* bo = (const float*)d_in[12];
  float* out = (float*)d_out;

  const size_t nAct = (size_t)kB * kT * kE;  // 4M elems per activation tensor
  const size_t nW   = (size_t)kE * kE;       // 1M elems per weight matrix

  // f16 workspace layout (all offsets in f16 elements; 64MB total)
  _Float16* p = (_Float16*)d_ws;
  _Float16* qx  = p;              p += nAct;   // query f16
  _Float16* kx  = p;              p += nAct;   // key f16
  _Float16* vx  = p;              p += nAct;   // value f16
  _Float16* wqh = p;              p += nW;
  _Float16* wkh = p;              p += nW;
  _Float16* wvh = p;              p += nW;
  _Float16* woh = p;              p += nW;
  _Float16* qws = p;              p += nAct;   // projected Q [B*H,T,D]
  _Float16* kws = p;              p += nAct;
  _Float16* vws = p;              p += nAct;
  _Float16* ows = p;              p += nAct;   // attention out [B,T,E]

  // 1) one-shot f32 -> f16 conversions
  const int actBlocks = (int)(nAct / 4 / 256);
  const int wBlocks   = (int)(nW / 4 / 256);
  cvt_kernel<<<actBlocks, 256, 0, stream>>>(query, qx, (int)nAct);
  cvt_kernel<<<actBlocks, 256, 0, stream>>>(key_,  kx, (int)nAct);
  cvt_kernel<<<actBlocks, 256, 0, stream>>>(value, vx, (int)nAct);
  cvt_kernel<<<wBlocks,   256, 0, stream>>>(wq, wqh, (int)nW);
  cvt_kernel<<<wBlocks,   256, 0, stream>>>(wk, wkh, (int)nW);
  cvt_kernel<<<wBlocks,   256, 0, stream>>>(wv, wvh, (int)nW);
  cvt_kernel<<<wBlocks,   256, 0, stream>>>(wo, woh, (int)nW);

  // 2) projections: grid = (row tiles, heads); 16x64 tile per wave
  const dim3 gGemm(kB * kT / 16, kH);  // (256, 16)
  proj_kernel<<<gGemm, dim3(32), 0, stream>>>(qx, wqh, bq, qws, 0.125f);
  proj_kernel<<<gGemm, dim3(32), 0, stream>>>(kx, wkh, bk, kws, 1.0f);
  proj_kernel<<<gGemm, dim3(32), 0, stream>>>(vx, wvh, bv, vws, 1.0f);

  // 3) flash attention (TDM-staged K/V)
  const dim3 gAttn(kB * kH, kT / 64);  // (64, 16), 4 waves per block
  attn_kernel<<<gAttn, dim3(128), 0, stream>>>(qws, kws, vws, mask, bias, ows);

  // 4) output projection
  outproj_kernel<<<gGemm, dim3(32), 0, stream>>>(ows, woh, bo, out);
}
